// MultiHeadAttention_33681133535386
// MI455X (gfx1250) — compile-verified, matrix operand order checked by reading, not run on hardware
//
#include <hip/hip_runtime.h>
#include <hip/hip_bf16.h>

// ---------------------------------------------------------------------------
// CDNA5 Tensor Data Mover availability (probe doc: builtin exists on both
// toolchains, 5-arg on ROCm 7.2 / clang-22, 6-arg on amdgpu-toolchain whose
// therock headers ship amd_gfx1250_TDM.h -- use that header as the arity probe)
// ---------------------------------------------------------------------------
#if defined(__has_builtin)
#  if __has_builtin(__builtin_amdgcn_tensor_load_to_lds) && \
      __has_builtin(__builtin_amdgcn_s_wait_tensorcnt)
#    define HAVE_TDM 1
#    if __has_include(<hip/amd_detail/amd_gfx1250_TDM.h>)
#      define TDM_6ARG 1
#    endif
#  endif
#endif
#ifndef HAVE_TDM
#  define HAVE_TDM 0
#endif

typedef _Float16 v16h __attribute__((ext_vector_type(16)));
typedef _Float16 h2   __attribute__((ext_vector_type(2)));
typedef float    v8f  __attribute__((ext_vector_type(8)));

#define HIDDEN 1024
#define NHEADS 16
#define HEADD  64
#define BATCH  2
#define SEQ    2048
#define MROWS  (BATCH * SEQ)   // 4096

#if HAVE_TDM
typedef unsigned int u32x4_t __attribute__((ext_vector_type(4)));
typedef int          i32x4_t __attribute__((ext_vector_type(4)));
typedef int          i32x8_t __attribute__((ext_vector_type(8)));

// ---------------------------------------------------------------------------
// Issue one TDM 2-D tile load (global -> LDS) per CDNA5 ISA.
// tile_d0 = contiguous elements per row, tile_d1 = rows, stride_elems = global
// row pitch (elements). LDS pitch = tile row + pad: pad_interval code selects
// 2^(code+1) dwords of data between pads, pad_amount code = (code+1) dwords.
// data_size = 2 bytes (f16). Tracked by TENSORcnt.
// ---------------------------------------------------------------------------
__device__ __forceinline__ void tdm_load_2d(unsigned lds_addr, const void* gsrc,
                                            unsigned tile_d0, unsigned tile_d1,
                                            unsigned stride_elems,
                                            unsigned pad_int_code,
                                            unsigned pad_amt_code) {
  const unsigned long long ga = (unsigned long long)(uintptr_t)gsrc;
  u32x4_t g0;
  g0[0] = 1u;                                    // count=1, user descriptor
  g0[1] = lds_addr;                              // LDS byte address
  g0[2] = (unsigned)ga;                          // global_addr[31:0]
  g0[3] = (unsigned)((ga >> 32) & 0x01FFFFFFull) // global_addr[56:32]
          | 0x80000000u;                         // type=2 ("image")
  i32x8_t g1;
  g1[0] = (int)((1u << 16)                       // data_size=1 -> 2 bytes
                | (1u << 20)                     // pad_enable
                | (pad_int_code << 22) | (pad_amt_code << 25));
  g1[1] = (int)((tile_d0 & 0xFFFFu) << 16);      // tensor_dim0[15:0] @bit48
  g1[2] = (int)(((tile_d0 >> 16) & 0xFFFFu) | ((tile_d1 & 0xFFFFu) << 16));
  g1[3] = (int)(((tile_d1 >> 16) & 0xFFFFu) | ((tile_d0 & 0xFFFFu) << 16)); // tile_dim0
  g1[4] = (int)(tile_d1 & 0xFFFFu);              // tile_dim1, tile_dim2=0 (2D)
  g1[5] = (int)stride_elems;                     // tensor_dim0_stride[31:0]
  g1[6] = 0;                                     // stride hi / dim1_stride (2D)
  g1[7] = 0;
  i32x4_t z4 = {0, 0, 0, 0};
#if defined(TDM_6ARG)
  i32x8_t z8 = {0, 0, 0, 0, 0, 0, 0, 0};
  __builtin_amdgcn_tensor_load_to_lds(g0, g1, z4, z4, z8, 0);
#else
  __builtin_amdgcn_tensor_load_to_lds(g0, g1, z4, z4, 0);
#endif
}
#endif  // HAVE_TDM

// ---------------------------------------------------------------------------
// 16x32 f16 fragment in the CDNA5 WMMA A/B VGPR layout (CDNA5 ISA):
// lane r=L%16 -> row (M or N), h=L/16 -> +8 K offset; VGPR pair v holds
// K=(v<4?2v:2v+8)+8h and K+1. Pairs contiguous -> dword / b128 DS reads.
// ---------------------------------------------------------------------------
__device__ __forceinline__ v16h load_frag(const _Float16* p, int row, int ld) {
  const int lane = threadIdx.x & 31;
  const int r = lane & 15;
  const int h = lane >> 4;
  const h2* rp = (const h2*)(p + (row + r) * ld + 8 * h);
  v16h out;
#pragma unroll
  for (int v = 0; v < 8; ++v) {
    const int k = (v < 4) ? (2 * v) : (2 * v + 8);
    h2 pr = rp[k >> 1];
    out[2 * v]     = pr.x;
    out[2 * v + 1] = pr.y;
  }
  return out;
}

__device__ __forceinline__ v8f wmma_f16(v16h a, v16h b, v8f c) {
  return __builtin_amdgcn_wmma_f32_16x16x32_f16(false, a, false, b,
                                                (short)0, c, false, false);
}

// ---------------------------------------------------------------------------
// fp32 -> f16 conversion (X)
// ---------------------------------------------------------------------------
__global__ void cvt_f32_to_f16(const float* __restrict__ in,
                               _Float16* __restrict__ out, int n) {
  int i = blockIdx.x * blockDim.x + threadIdx.x;
  if (i < n) out[i] = (_Float16)in[i];
}

// ---------------------------------------------------------------------------
// fp32 [R][C] -> f16 transposed [C][R] (weights), tiled through LDS.
// block (32,8), grid (C/32, R/32).
// ---------------------------------------------------------------------------
__global__ void cvt_transpose_f16(const float* __restrict__ in,
                                  _Float16* __restrict__ out, int R, int C) {
  __shared__ float tile[32][33];
  const int c0 = blockIdx.x * 32, r0 = blockIdx.y * 32;
  const int tx = threadIdx.x, ty = threadIdx.y;
#pragma unroll
  for (int i = 0; i < 32; i += 8)
    tile[ty + i][tx] = in[(size_t)(r0 + ty + i) * C + c0 + tx];
  __syncthreads();
#pragma unroll
  for (int i = 0; i < 32; i += 8)
    out[(size_t)(c0 + ty + i) * R + r0 + tx] = (_Float16)tile[tx][ty + i];
}

// ---------------------------------------------------------------------------
// C[M,N] = A[M,K] @ Wt[N,K]^T + bias; out = (acc+bias)*scale.
// 256 threads = 8 waves, 128x128 tile, BK=32, double-buffered LDS staged by
// the TDM (pad feature produces the 40-half pitch) or uint4 copies.
// OUT_TRANS stores the result transposed ([N][M]) for the V projection.
// ---------------------------------------------------------------------------
template <bool OUT_HALF, bool OUT_TRANS>
__launch_bounds__(256)
__global__ void gemm_bias_kernel(const _Float16* __restrict__ A,
                                 const _Float16* __restrict__ Wt,
                                 const float* __restrict__ bias,
                                 void* __restrict__ outp, float scale,
                                 int M, int N, int K) {
  const int LD = 40;                        // 32 + 8 halves pad
  __shared__ _Float16 Ash[2][128 * LD];
  __shared__ _Float16 Bsh[2][128 * LD];     // [n][k]

  const int tid  = threadIdx.x;
  const int lane = tid & 31;
  const int wid  = tid >> 5;
  const int wm   = wid & 3;
  const int wn   = wid >> 2;
  const int blockN = blockIdx.x * 128;
  const int blockM = blockIdx.y * 128;

  auto stage = [&](int buf, int k0) {
#if HAVE_TDM
    if (wid == 0) {
      tdm_load_2d((unsigned)(uintptr_t)&Ash[buf][0],
                  A + (size_t)blockM * K + k0, 32, 128, (unsigned)K, 3, 3);
      tdm_load_2d((unsigned)(uintptr_t)&Bsh[buf][0],
                  Wt + (size_t)blockN * K + k0, 32, 128, (unsigned)K, 3, 3);
    }
#else
    const int row = tid >> 1;
    const int ch  = (tid & 1) * 16;
    {
      const uint4* s = (const uint4*)(A + (size_t)(blockM + row) * K + k0 + ch);
      uint4* d = (uint4*)(&Ash[buf][0] + row * LD + ch);
      d[0] = s[0]; d[1] = s[1];
    }
    {
      const uint4* s = (const uint4*)(Wt + (size_t)(blockN + row) * K + k0 + ch);
      uint4* d = (uint4*)(&Bsh[buf][0] + row * LD + ch);
      d[0] = s[0]; d[1] = s[1];
    }
#endif
  };
  auto wait_stage = [&]() {
#if HAVE_TDM
    __builtin_amdgcn_s_wait_tensorcnt(0);
#endif
    __syncthreads();
  };

  v8f acc[2][4];
  const v8f vzero = {};
#pragma unroll
  for (int mi = 0; mi < 2; ++mi)
#pragma unroll
    for (int ni = 0; ni < 4; ++ni) acc[mi][ni] = vzero;

  stage(0, 0);
  wait_stage();
  const int NITER = K / 32;
  for (int it = 0; it < NITER; ++it) {
    const int cur = it & 1;
    if (it + 1 < NITER) stage(cur ^ 1, (it + 1) * 32);  // overlap with WMMA

    const _Float16* Ab = &Ash[cur][0];
    const _Float16* Bb = &Bsh[cur][0];
    // Batch ALL fragment loads before the WMMA burst so the DS loads issue
    // back-to-back and the scheduler can use partial s_wait_dscnt counts.
    v16h af[2], bf[4];
    af[0] = load_frag(Ab, wm * 32 + 0, LD);
    af[1] = load_frag(Ab, wm * 32 + 16, LD);
#pragma unroll
    for (int ni = 0; ni < 4; ++ni) bf[ni] = load_frag(Bb, wn * 64 + ni * 16, LD);
#pragma unroll
    for (int ni = 0; ni < 4; ++ni)
#pragma unroll
      for (int mi = 0; mi < 2; ++mi)
        acc[mi][ni] = wmma_f16(af[mi], bf[ni], acc[mi][ni]);
    wait_stage();
  }

  // epilogue: C/D layout (lane%16 = N, elem v + 8*(lane/16) = M)
  const int r = lane & 15;
  const int h = lane >> 4;
#pragma unroll
  for (int ni = 0; ni < 4; ++ni) {
    const int col = blockN + wn * 64 + ni * 16 + r;
    const float bv = bias[col];
#pragma unroll
    for (int mi = 0; mi < 2; ++mi) {
      const int row0 = blockM + wm * 32 + mi * 16 + 8 * h;
#pragma unroll
      for (int v = 0; v < 8; ++v) {
        const float val = (acc[mi][ni][v] + bv) * scale;
        if (OUT_TRANS) {  // [N][M]; consecutive v are contiguous -> merged
          ((_Float16*)outp)[(size_t)col * M + row0 + v] = (_Float16)val;
        } else if (OUT_HALF) {
          ((_Float16*)outp)[(size_t)(row0 + v) * N + col] = (_Float16)val;
        } else {
          ((float*)outp)[(size_t)(row0 + v) * N + col] = val;
        }
      }
    }
  }
}

// ---------------------------------------------------------------------------
// Flash attention: grid = B*NHEADS*(S/64), 128 threads = 4 waves, each wave
// owns 16 query rows. K (row-major) and Vt (pre-transposed [dim][row]) chunks
// of 32 keys are double-buffered in LDS via TDM / copies. Q pre-scaled 1/8.
// ---------------------------------------------------------------------------
__launch_bounds__(128)
__global__ void flash_attn_kernel(const _Float16* __restrict__ Q,
                                  const _Float16* __restrict__ Km,
                                  const _Float16* __restrict__ Vt,
                                  _Float16* __restrict__ Ctx) {
  const int LDK = HEADD + 8;                // 72-half pitch (TDM pad 32dw+4dw)
  const int LDP = 32 + 8;                   // 40-half pitch (TDM pad 16dw+4dw)
  __shared__ _Float16 Ksh[2][32 * LDK];     // [key][dim]
  __shared__ _Float16 Vsh[2][HEADD * LDP];  // [dim][key]
  __shared__ _Float16 Psh[4 * 16 * LDP];    // per-wave probs [row][key]

  const int tid  = threadIdx.x;
  const int lane = tid & 31;
  const int wid  = tid >> 5;
  const int r = lane & 15;
  const int h = lane >> 4;

  const int qblk = blockIdx.x & 31;
  const int head = (blockIdx.x >> 5) & (NHEADS - 1);
  const int b    = blockIdx.x >> 9;

  const int qrow0   = b * SEQ + qblk * 64 + wid * 16;
  const int colbase = head * HEADD;

  auto stage = [&](int buf, int kc) {
#if HAVE_TDM
    if (wid == 0) {
      tdm_load_2d((unsigned)(uintptr_t)&Ksh[buf][0],
                  Km + (size_t)(b * SEQ + kc) * HIDDEN + colbase,
                  64, 32, HIDDEN, 4, 3);
      tdm_load_2d((unsigned)(uintptr_t)&Vsh[buf][0],
                  Vt + (size_t)colbase * MROWS + b * SEQ + kc,
                  32, 64, MROWS, 3, 3);
    }
#else
    {
      const int key = tid >> 2;
      const int db  = (tid & 3) * 16;
      const uint4* s = (const uint4*)(Km + (size_t)(b * SEQ + kc + key) * HIDDEN +
                                      colbase + db);
      uint4* d = (uint4*)(&Ksh[buf][0] + key * LDK + db);
      d[0] = s[0]; d[1] = s[1];
    }
    {
      const int dim = tid >> 1;
      const int kb  = (tid & 1) * 16;
      const uint4* s = (const uint4*)(Vt + (size_t)(colbase + dim) * MROWS +
                                      b * SEQ + kc + kb);
      uint4* d = (uint4*)(&Vsh[buf][0] + dim * LDP + kb);
      d[0] = s[0]; d[1] = s[1];
    }
#endif
  };
  auto wait_stage = [&]() {
#if HAVE_TDM
    __builtin_amdgcn_s_wait_tensorcnt(0);
#endif
    __syncthreads();
  };

  // Q fragments (A-layout) direct from global, one per 32-wide K-step
  v16h qf[2];
  qf[0] = load_frag(Q + colbase + 0,  qrow0, HIDDEN);
  qf[1] = load_frag(Q + colbase + 32, qrow0, HIDDEN);

  float m_[8], l_[8];
  v8f ctxacc[4];
  const v8f vzero = {};
#pragma unroll
  for (int v = 0; v < 8; ++v) { m_[v] = -1e30f; l_[v] = 0.f; }
#pragma unroll
  for (int ni = 0; ni < 4; ++ni) ctxacc[ni] = vzero;

  stage(0, 0);
  wait_stage();
  const int NITER = SEQ / 32;
  for (int it = 0; it < NITER; ++it) {
    const int cur = it & 1;
    if (it + 1 < NITER) stage(cur ^ 1, (it + 1) * 32);

    const _Float16* Kb = &Ksh[cur][0];
    const _Float16* Vb = &Vsh[cur][0];

    // scores S = Q(16x64) @ K^T : batch the 4 K fragments, then 4 WMMAs
    v16h kf[2][2];
#pragma unroll
    for (int t = 0; t < 2; ++t) {
      kf[t][0] = load_frag(Kb + 0,  t * 16, LDK);
      kf[t][1] = load_frag(Kb + 32, t * 16, LDK);
    }
    v8f s[2];
#pragma unroll
    for (int t = 0; t < 2; ++t) {
      v8f a = vzero;
      a = wmma_f16(qf[0], kf[t][0], a);
      a = wmma_f16(qf[1], kf[t][1], a);
      s[t] = a;
    }

    // online softmax (row lives across the 16 lanes of one half)
    float corr[8];
    _Float16* pw = Psh + wid * 16 * LDP;
#pragma unroll
    for (int v = 0; v < 8; ++v) {
      float rm = fmaxf(s[0][v], s[1][v]);
#pragma unroll
      for (int msk = 1; msk < 16; msk <<= 1)
        rm = fmaxf(rm, __shfl_xor(rm, msk, 32));
      const float nm = fmaxf(m_[v], rm);
      const float p0 = __expf(s[0][v] - nm);
      const float p1 = __expf(s[1][v] - nm);
      float rs = p0 + p1;
#pragma unroll
      for (int msk = 1; msk < 16; msk <<= 1) rs += __shfl_xor(rs, msk, 32);
      corr[v] = __expf(m_[v] - nm);
      l_[v] = l_[v] * corr[v] + rs;
      m_[v] = nm;
      const int row = v + 8 * h;
      pw[row * LDP + r]      = (_Float16)p0;
      pw[row * LDP + 16 + r] = (_Float16)p1;
    }

    // ctx = P(16x32) @ V(32x64) + ctx*corr : batch P + 4 V frags, then WMMAs
    v16h pf = load_frag(pw, 0, LDP);
    v16h vf[4];
#pragma unroll
    for (int ni = 0; ni < 4; ++ni) vf[ni] = load_frag(Vb, ni * 16, LDP);
#pragma unroll
    for (int ni = 0; ni < 4; ++ni) {
      v8f c;
#pragma unroll
      for (int v = 0; v < 8; ++v) c[v] = ctxacc[ni][v] * corr[v];
      ctxacc[ni] = wmma_f16(pf, vf[ni], c);
    }
    wait_stage();
  }

  // normalize and write context (f16)
#pragma unroll
  for (int ni = 0; ni < 4; ++ni) {
    const int col = colbase + ni * 16 + r;
#pragma unroll
    for (int v = 0; v < 8; ++v) {
      const int row = qrow0 + v + 8 * h;
      Ctx[(size_t)row * HIDDEN + col] = (_Float16)(ctxacc[ni][v] / l_[v]);
    }
  }
}

// ---------------------------------------------------------------------------
// Launcher
// ---------------------------------------------------------------------------
extern "C" void kernel_launch(void* const* d_in, const int* in_sizes, int n_in,
                              void* d_out, int out_size, void* d_ws,
                              size_t ws_size, hipStream_t stream) {
  (void)in_sizes; (void)n_in; (void)out_size; (void)ws_size;
  const float* x  = (const float*)d_in[0];
  const float* Wq = (const float*)d_in[1];
  const float* bq = (const float*)d_in[2];
  const float* Wk = (const float*)d_in[3];
  const float* bk = (const float*)d_in[4];
  const float* Wv = (const float*)d_in[5];
  const float* bv = (const float*)d_in[6];
  const float* Wo = (const float*)d_in[7];
  const float* bo = (const float*)d_in[8];

  const size_t nX = (size_t)MROWS * HIDDEN;
  const size_t nW = (size_t)HIDDEN * HIDDEN;

  _Float16* Xh  = (_Float16*)d_ws;
  _Float16* Wqt = Xh + nX;      // transposed f16 weights [N][K]
  _Float16* Wkt = Wqt + nW;
  _Float16* Wvt = Wkt + nW;
  _Float16* Wot = Wvt + nW;
  _Float16* Qh  = Wot + nW;
  _Float16* Kh  = Qh + nX;
  _Float16* Vtr = Kh + nX;      // V stored transposed [HIDDEN][MROWS]
  _Float16* Ch  = Xh;           // reuse X buffer for attention output

  const int T = 256;
  cvt_f32_to_f16<<<(int)((nX + T - 1) / T), T, 0, stream>>>(x, Xh, (int)nX);
  dim3 tb(32, 8), tg(HIDDEN / 32, HIDDEN / 32);
  cvt_transpose_f16<<<tg, tb, 0, stream>>>(Wq, Wqt, HIDDEN, HIDDEN);
  cvt_transpose_f16<<<tg, tb, 0, stream>>>(Wk, Wkt, HIDDEN, HIDDEN);
  cvt_transpose_f16<<<tg, tb, 0, stream>>>(Wv, Wvt, HIDDEN, HIDDEN);
  cvt_transpose_f16<<<tg, tb, 0, stream>>>(Wo, Wot, HIDDEN, HIDDEN);

  dim3 gg(HIDDEN / 128, MROWS / 128);
  const float qscale = 0.125f;  // 1/sqrt(HEAD_DIM) folded into Q projection
  gemm_bias_kernel<true, false><<<gg, 256, 0, stream>>>(
      Xh, Wqt, bq, Qh, qscale, MROWS, HIDDEN, HIDDEN);
  gemm_bias_kernel<true, false><<<gg, 256, 0, stream>>>(
      Xh, Wkt, bk, Kh, 1.0f, MROWS, HIDDEN, HIDDEN);
  gemm_bias_kernel<true, true><<<gg, 256, 0, stream>>>(
      Xh, Wvt, bv, Vtr, 1.0f, MROWS, HIDDEN, HIDDEN);

  flash_attn_kernel<<<BATCH * NHEADS * (SEQ / 64), 128, 0, stream>>>(Qh, Kh,
                                                                     Vtr, Ch);

  gemm_bias_kernel<false, false><<<gg, 256, 0, stream>>>(
      Ch, Wot, bo, d_out, 1.0f, MROWS, HIDDEN, HIDDEN);
}
